// CTCDecoderN_82111184765030
// MI455X (gfx1250) — matrix operand order repaired and checked by reading, not compiled, not run on hardware
//
#include <hip/hip_runtime.h>
#include <hip/hip_bf16.h>

#define LL 4096
#define BB 4
#define DD 512
#define VV 1000
#define VP 1024

typedef __bf16 bf16;
typedef __attribute__((ext_vector_type(16))) __bf16 v16bf;
typedef __attribute__((ext_vector_type(8)))  float  v8f;
typedef __attribute__((ext_vector_type(4)))  unsigned int v4u;
typedef __attribute__((ext_vector_type(8)))  int v8i;
typedef __attribute__((ext_vector_type(4)))  int v4i;

// AS-qualified pointee types for the async-to-LDS builtin
// (addrspace(1) prints as __device__, addrspace(3) as __shared__ in HIP diags)
typedef __attribute__((address_space(1))) v4i g_v4i;
typedef __attribute__((address_space(3))) v4i l_v4i;
typedef __attribute__((address_space(3))) char lds_char;

// ---------------------------------------------------------------------------
// WMMA helpers (CDNA5 wave32, V_WMMA_F32_16X16X32_BF16)
// ---------------------------------------------------------------------------
__device__ __forceinline__ v8f wmma_bf16(v16bf a, v16bf b, v8f c) {
  return __builtin_amdgcn_wmma_f32_16x16x32_bf16(false, a, false, b, (short)0, c,
                                                 false, false);
}

// A-matrix 16x32 bf16 (MxK), source row-major [row][k], stride ld elements.
__device__ __forceinline__ v16bf frag_a(const bf16* base, int ld, int lane) {
  int row = lane & 15;
  int kb  = (lane >> 4) << 4;
  const bf16* p0 = base + row * ld + kb;
  const bf16* p1 = p0 + 8;
  v16bf a;
#pragma unroll
  for (int i = 0; i < 8; ++i) { a[i] = p0[i]; a[8 + i] = p1[i]; }
  return a;
}

// B-matrix 32x16 bf16 (KxN), source stored as [n][k] row-major (B^T), stride ld.
__device__ __forceinline__ v16bf frag_b(const bf16* base, int ld, int lane) {
  int n  = lane & 15;
  int kb = (lane >> 4) << 4;
  const bf16* p = base + n * ld + kb;
  v16bf b;
#pragma unroll
  for (int i = 0; i < 16; ++i) b[i] = p[i];
  return b;
}

// ---------------------------------------------------------------------------
// CDNA5 data movers
// ---------------------------------------------------------------------------
// Per-lane 16B async copy global -> LDS (tracked by ASYNCcnt)
__device__ __forceinline__ void async_copy16(void* ldst, const void* gsrc) {
  __builtin_amdgcn_global_load_async_to_lds_b128((g_v4i*)gsrc, (l_v4i*)ldst,
                                                 0, 0);
}

// TDM: DMA a 2D tile (rows x 512 bf16, row-major, stride 512) global -> LDS.
// Descriptor per CDNA5 ISA ch.8 (D# group0/group1). Issue once per wave.
__device__ __forceinline__ void tdm_load_rows512(void* lds_dst, const bf16* gsrc,
                                                 unsigned rows) {
  unsigned lofs = (unsigned)(size_t)(lds_char*)lds_dst;       // LDS byte address
  unsigned long long ga = (unsigned long long)(size_t)gsrc;   // global byte addr
  v4u g0;
  g0[0] = 1u;                                        // count=1 valid descriptor
  g0[1] = lofs;                                      // lds_addr
  g0[2] = (unsigned)(ga & 0xffffffffull);            // global_addr[31:0]
  g0[3] = (unsigned)((ga >> 32) & 0x01ffffffull)     // global_addr[56:32]
          | (2u << 30);                              // type = 2 (image)
  v8i g1;
  g1[0] = (int)(1u << 16);          // wg_mask=0, data_size=1 (2 bytes)
  g1[1] = (int)(512u << 16);        // tensor_dim0[15:0] = 512
  g1[2] = (int)(rows << 16);        // tensor_dim0[31:16]=0, tensor_dim1[15:0]=rows
  g1[3] = (int)(512u << 16);        // tensor_dim1[31:16]=0, tile_dim0 = 512
  g1[4] = (int)rows;                // tile_dim1 = rows, tile_dim2 = 0
  g1[5] = (int)512u;                // tensor_dim0_stride[31:0] = 512
  g1[6] = 0;                        // stride0[47:32] = 0, stride1[15:0] = 0
  g1[7] = 0;                        // stride1[47:16] = 0
  v4i z4 = {};
  v8i z8 = {};
  __builtin_amdgcn_tensor_load_to_lds(g0, g1, z4, z4, z8, 0);
}

// ---------------------------------------------------------------------------
// Packing kernels (f32 -> bf16, with transpose / padding)
// ---------------------------------------------------------------------------
__global__ void k_pack_x(const float* __restrict__ x, bf16* __restrict__ xbf) {
  int idx = blockIdx.x * blockDim.x + threadIdx.x;  // over (l,b,d)
  int d = idx & (DD - 1);
  int t = idx >> 9;
  int b = t & (BB - 1);
  int l = t >> 2;
  xbf[(b * LL + l) * DD + d] = (bf16)x[idx];
}

__global__ void k_pack_ww(const float* __restrict__ w, bf16* __restrict__ wb) {
  int idx = blockIdx.x * blockDim.x + threadIdx.x;
  wb[idx] = (bf16)w[idx];
}

__global__ void k_pack_lw(const float* __restrict__ lw, bf16* __restrict__ lwbf) {
  int idx = blockIdx.x * blockDim.x + threadIdx.x;  // over VP*DD
  int v = idx >> 9;
  lwbf[idx] = (v < VV) ? (bf16)lw[idx] : (bf16)0.0f;
}

// ---------------------------------------------------------------------------
// wx = xb @ Ww^T + Wb   (rows flattened over B*L), output bf16
// ---------------------------------------------------------------------------
__global__ void __launch_bounds__(256)
k_gemm_wx(const bf16* __restrict__ xbf, const bf16* __restrict__ wwbf,
          const float* __restrict__ Wb, bf16* __restrict__ wxbf) {
  int lane = threadIdx.x & 31;
  int wave = threadIdx.x >> 5;
  int r0 = blockIdx.x * 64;
  int c0 = blockIdx.y * 64;
  int mi  = wave >> 1;
  int nb0 = (wave & 1) * 2;
  v8f vzero = {};
  v8f acc0 = vzero, acc1 = vzero;
  const bf16* abase  = xbf + (r0 + mi * 16) * DD;
  const bf16* b0base = wwbf + (c0 + nb0 * 16) * DD;
  const bf16* b1base = b0base + 16 * DD;
  for (int k = 0; k < DD; k += 32) {
    v16bf a = frag_a(abase + k, DD, lane);
    acc0 = wmma_bf16(a, frag_b(b0base + k, DD, lane), acc0);
    acc1 = wmma_bf16(a, frag_b(b1base + k, DD, lane), acc1);
  }
  int n  = lane & 15;
  int mo = (lane >> 4) << 3;
  int row = r0 + mi * 16 + mo;
  int cA = c0 + nb0 * 16 + n;
  int cB = cA + 16;
  float biasA = Wb[cA], biasB = Wb[cB];
#pragma unroll
  for (int i = 0; i < 8; ++i) {
    wxbf[(row + i) * DD + cA] = (bf16)(acc0[i] + biasA);
    wxbf[(row + i) * DD + cB] = (bf16)(acc1[i] + biasB);
  }
}

// ---------------------------------------------------------------------------
// Fused: Y[b, l0:l0+32, :] = sum_m (xl @ wx_m^T) @ x_m     (flash-style)
// xl staged via async-to-LDS (ASYNCcnt); wx tile streamed via TDM (TENSORcnt)
// ---------------------------------------------------------------------------
__global__ void __launch_bounds__(256)
k_fused_y(const bf16* __restrict__ xbf, const bf16* __restrict__ wxbf,
          bf16* __restrict__ ybf) {
  __shared__ __align__(16) bf16 sXl[32 * DD];    // 32 KB  A rows (l-tile)
  __shared__ __align__(16) bf16 sWx[64 * DD];    // 64 KB  wx m-tile, [m][k]
  __shared__ __align__(16) bf16 sXmT[DD * 64];   // 64 KB  x m-tile transposed [d][m]
  __shared__ __align__(16) bf16 sS[32 * 64];     //  4 KB  S tile (bf16)
  int lane = threadIdx.x & 31;
  int wave = threadIdx.x >> 5;
  int b  = blockIdx.y;
  int l0 = blockIdx.x * 32;
  const bf16* xb  = xbf  + b * LL * DD;
  const bf16* wxb = wxbf + b * LL * DD;

  {  // stage xl once: per-lane async 16B copies, tracked by ASYNCcnt
    const uint4* src = (const uint4*)(xb + l0 * DD);
    uint4* dst = (uint4*)sXl;
    for (int i = threadIdx.x; i < 32 * DD / 8; i += 256)
      async_copy16(dst + i, src + i);
    __builtin_amdgcn_s_wait_asynccnt(0);
  }

  v8f vzero = {};
  v8f acc[8];
#pragma unroll
  for (int j = 0; j < 8; ++j) acc[j] = vzero;

  int ry  = wave & 1;          // Y row-block (also S row-block)
  int cb0 = (wave >> 1) * 8;   // Y col-block base
  int sx  = wave >> 1;         // S col-block

  for (int m0 = 0; m0 < LL; m0 += 64) {
    __syncthreads();
    // TDM: wave 0 DMAs the 64x512 wx tile straight into LDS
    if (wave == 0) tdm_load_rows512(sWx, wxb + m0 * DD, 64u);
    // prefetch next iteration's x tile while we shuffle this one
    if (m0 + 64 < LL) __builtin_prefetch(xb + (m0 + 64) * DD, 0, 1);
    {  // stream x tile transposed -> [d][m] (manual scatter; TDM can't transpose)
      for (int i = threadIdx.x; i < 64 * DD / 8; i += 256) {
        int mm = i >> 6;
        int dc = (i & 63) << 3;
        const bf16* src = xb + (m0 + mm) * DD + dc;
#pragma unroll
        for (int e = 0; e < 8; ++e) sXmT[(dc + e) * 64 + mm] = src[e];
      }
    }
    if (wave == 0) __builtin_amdgcn_s_wait_tensorcnt(0);
    __syncthreads();

    // S(32x64) = xl @ wx_m^T ; one 16x16 subtile per wave, K=512
    v8f s = vzero;
    const bf16* sa = sXl + (ry * 16) * DD;
    const bf16* sb = sWx + (sx * 16) * DD;
    for (int k = 0; k < DD; k += 32)
      s = wmma_bf16(frag_a(sa + k, DD, lane), frag_b(sb + k, DD, lane), s);
    {  // spill S subtile to LDS as bf16 (A-operand of second GEMM)
      int n  = lane & 15;
      int mo = (lane >> 4) << 3;
#pragma unroll
      for (int i = 0; i < 8; ++i)
        sS[(ry * 16 + mo + i) * 64 + sx * 16 + n] = (bf16)s[i];
    }
    __syncthreads();

    // Y(32x512) += S @ x_m ; K=64
#pragma unroll
    for (int kk = 0; kk < 64; kk += 32) {
      v16bf aS = frag_a(sS + (ry * 16) * 64 + kk, 64, lane);
#pragma unroll
      for (int j = 0; j < 8; ++j)
        acc[j] = wmma_bf16(aS, frag_b(sXmT + ((cb0 + j) * 16) * 64 + kk, 64, lane),
                           acc[j]);
    }
  }

  // write Y tile (bf16)
  int n  = lane & 15;
  int mo = (lane >> 4) << 3;
  bf16* yb = ybf + (b * LL + l0 + ry * 16 + mo) * DD;
#pragma unroll
  for (int j = 0; j < 8; ++j) {
    int d = (cb0 + j) * 16 + n;
#pragma unroll
    for (int i = 0; i < 8; ++i) yb[i * DD + d] = (bf16)acc[j][i];
  }
}

// ---------------------------------------------------------------------------
// logits = y @ Lw^T + Lb, then log_softmax over V, write (L,B,V) f32
// ---------------------------------------------------------------------------
__global__ void __launch_bounds__(256)
k_logits_lsm(const bf16* __restrict__ ybf, const bf16* __restrict__ lwbf,
             const float* __restrict__ Lb, float* __restrict__ out) {
  __shared__ __align__(16) float sL[16 * VP];  // 64 KB logits tile
  __shared__ float sRed[16][16];
  __shared__ float sRowMax[16];
  __shared__ float sRowSum[16];
  int lane = threadIdx.x & 31;
  int wave = threadIdx.x >> 5;
  int r0 = blockIdx.x * 16;  // row over flattened (B,L)
  const bf16* abase = ybf + r0 * DD;

  v8f vzero = {};
  v8f acc[8];
#pragma unroll
  for (int j = 0; j < 8; ++j) acc[j] = vzero;

  for (int k = 0; k < DD; k += 32) {
    v16bf a = frag_a(abase + k, DD, lane);
#pragma unroll
    for (int j = 0; j < 8; ++j) {
      int cb = wave * 8 + j;
      acc[j] = wmma_bf16(a, frag_b(lwbf + cb * 16 * DD + k, DD, lane), acc[j]);
    }
  }
  {
    int n  = lane & 15;
    int mo = (lane >> 4) << 3;
#pragma unroll
    for (int j = 0; j < 8; ++j) {
      int v = (wave * 8 + j) * 16 + n;
      float bias = (v < VV) ? Lb[v] : 0.0f;
#pragma unroll
      for (int i = 0; i < 8; ++i)
        sL[(mo + i) * VP + v] = (v < VV) ? (acc[j][i] + bias) : -1e30f;
    }
  }
  __syncthreads();

  int row = threadIdx.x >> 4;
  int sub = threadIdx.x & 15;
  float m = -1e30f;
  for (int v = sub; v < VV; v += 16) m = fmaxf(m, sL[row * VP + v]);
  sRed[row][sub] = m;
  __syncthreads();
  if (sub == 0) {
    float mm = sRed[row][0];
    for (int i = 1; i < 16; ++i) mm = fmaxf(mm, sRed[row][i]);
    sRowMax[row] = mm;
  }
  __syncthreads();
  float rmax = sRowMax[row];
  float ssum = 0.f;
  for (int v = sub; v < VV; v += 16) ssum += __expf(sL[row * VP + v] - rmax);
  sRed[row][sub] = ssum;
  __syncthreads();
  if (sub == 0) {
    float t = 0.f;
    for (int i = 0; i < 16; ++i) t += sRed[row][i];
    sRowSum[row] = t;
  }
  __syncthreads();
  float lse = rmax + __logf(sRowSum[row]);
  int gr = r0 + row;
  int b = gr >> 12;         // / LL
  int l = gr & (LL - 1);
  float* orow = out + (l * BB + b) * VV;
  for (int v = sub; v < VV; v += 16) orow[v] = sL[row * VP + v] - lse;
}

// ---------------------------------------------------------------------------
extern "C" void kernel_launch(void* const* d_in, const int* in_sizes, int n_in,
                              void* d_out, int out_size, void* d_ws, size_t ws_size,
                              hipStream_t stream) {
  const float* x  = (const float*)d_in[0];
  const float* Ww = (const float*)d_in[1];
  const float* Wb = (const float*)d_in[2];
  const float* Lw = (const float*)d_in[3];
  const float* Lb = (const float*)d_in[4];
  float* out = (float*)d_out;
  char* ws = (char*)d_ws;

  const size_t SZ = (size_t)BB * LL * DD * 2;  // 16 MiB per bf16 tensor
  bf16* xbf  = (bf16*)(ws);
  bf16* wxbf = (bf16*)(ws + SZ);
  bf16* ybf  = (bf16*)(ws + 2 * SZ);
  bf16* wwbf = (bf16*)(ws + 3 * SZ);
  bf16* lwbf = (bf16*)(ws + 3 * SZ + (size_t)DD * DD * 2);

  k_pack_x <<<(LL * BB * DD) / 256, 256, 0, stream>>>(x, xbf);
  k_pack_ww<<<(DD * DD) / 256,      256, 0, stream>>>(Ww, wwbf);
  k_pack_lw<<<(VP * DD) / 256,      256, 0, stream>>>(Lw, lwbf);
  k_gemm_wx<<<dim3((BB * LL) / 64, DD / 64), 256, 0, stream>>>(xbf, wwbf, Wb, wxbf);
  k_fused_y<<<dim3(LL / 32, BB),             256, 0, stream>>>(xbf, wxbf, ybf);
  k_logits_lsm<<<(BB * LL) / 16,             256, 0, stream>>>(ybf, lwbf, Lb, out);
}
